// ParallelTransformerBlock_29918742184225
// MI455X (gfx1250) — compile-verified, matrix operand order checked by reading, not run on hardware
//
#include <hip/hip_runtime.h>
#include <hip/hip_bf16.h>

// ---------------- types ----------------
typedef __bf16 bf16;
typedef __bf16 v16bf __attribute__((ext_vector_type(16)));
typedef __bf16 v8bf  __attribute__((ext_vector_type(8)));
typedef float  v8f   __attribute__((ext_vector_type(8)));
typedef int    v4i   __attribute__((ext_vector_type(4)));

union FragAB { v16bf v; v8bf h[2]; };

// gfx1250 async memory->LDS path (guarded: falls back to sync staging)
#if defined(__gfx1250__) && __has_builtin(__builtin_amdgcn_global_load_async_to_lds_b128)
#define HAVE_ASYNC_LDS 1
typedef __attribute__((address_space(1))) v4i as1_v4i;   // global int4
typedef __attribute__((address_space(3))) v4i as3_v4i;   // LDS int4
#else
#define HAVE_ASYNC_LDS 0
#endif

__device__ __forceinline__ void wait_async_zero() {
#if HAVE_ASYNC_LDS
#if __has_builtin(__builtin_amdgcn_s_wait_asynccnt)
    __builtin_amdgcn_s_wait_asynccnt(0);
#else
    asm volatile("s_wait_asynccnt 0x0" ::: "memory");
#endif
#endif
}

// ---------------- model constants ----------------
#define DIMC        2048
#define HEADS       16
#define DIM_HEAD    128
#define ATTN_INNER  2048
#define FF_INNER    8192
#define NPROJ       18688          // 2048+128+128+8192+8192
#define SEQ         2048
#define BATCH       2
#define MROWS       (BATCH*SEQ)    // 4096
#define COL_K       2048
#define COL_V       2176
#define COL_FF      2304
#define COL_GATE    10496
#define EPSV        1e-5f
#define RMS_SCALE_F 45.254833995939045f   // sqrt(2048)
#define Q_SCALE_F   0.08838834764831845f  // 128^-0.5

// ---------------- WMMA fragment load: 16x32 bf16, rows contiguous in K ----------------
// Works for A (row-major [M][K]) and for B given as B^T (row-major [N][K]).
__device__ __forceinline__ v16bf load_frag16x32(const bf16* base, int ld) {
    const int lane = threadIdx.x & 31;
    const int sel  = lane >> 4;      // K-half select per ISA 16-bit A layout
    const int r    = lane & 15;      // M (or N) index
    const bf16* p = base + (size_t)r * (size_t)ld + sel * 8;
    FragAB f;
    f.h[0] = *reinterpret_cast<const v8bf*>(p);       // K = 0..7   (or 8..15)
    f.h[1] = *reinterpret_cast<const v8bf*>(p + 16);  // K = 16..23 (or 24..31)
    return f.v;
}

__device__ __forceinline__ v8f wmma_bf16(v16bf a, v16bf b, v8f c) {
    return __builtin_amdgcn_wmma_f32_16x16x32_bf16(false, a, false, b, (short)0, c,
                                                   false, false);
}

// ---------------- transpose + f32 -> bf16 (weights to N-major) ----------------
__global__ __launch_bounds__(256)
void transpose_cvt_kernel(const float* __restrict__ in, bf16* __restrict__ out,
                          int K, int N) {
    __shared__ float tile[32][33];
    const int nb = blockIdx.x * 32, kb = blockIdx.y * 32;
    const int tx = threadIdx.x, ty = threadIdx.y;   // 32 x 8
    #pragma unroll
    for (int i = 0; i < 32; i += 8)
        tile[ty + i][tx] = in[(size_t)(kb + ty + i) * N + nb + tx];
    __syncthreads();
    #pragma unroll
    for (int i = 0; i < 32; i += 8)
        out[(size_t)(nb + ty + i) * K + kb + tx] = (bf16)tile[tx][ty + i];
}

// ---------------- RMSNorm -> bf16 ----------------
__global__ __launch_bounds__(256)
void rmsnorm_kernel(const float* __restrict__ x, const float* __restrict__ gamma,
                    bf16* __restrict__ xn) {
    const int row = blockIdx.x;
    const int tid = threadIdx.x;
    __shared__ float red[256];
    const float* xr = x + (size_t)row * DIMC;
    float s = 0.f;
    for (int c = tid; c < DIMC; c += 256) { float v = xr[c]; s += v * v; }
    red[tid] = s; __syncthreads();
    for (int off = 128; off > 0; off >>= 1) {
        if (tid < off) red[tid] += red[tid + off];
        __syncthreads();
    }
    const float inv = rsqrtf(red[0] + EPSV) * RMS_SCALE_F;
    bf16* xo = xn + (size_t)row * DIMC;
    for (int c = tid; c < DIMC; c += 256)
        xo[c] = (bf16)(xr[c] * inv * gamma[c]);
}

// ---------------- bf16 WMMA GEMM: C[M,N] = A[M,K] * Bt[N,K]^T ----------------
// 128x128 block tile, 8 waves (4 x 2), each wave 32x64 (2x4 wmma tiles), K-step 32.
// Double-buffered LDS staging via GLOBAL_LOAD_ASYNC_TO_LDS_B128 (ASYNCcnt) when
// available; falls back to synchronous b128 copies otherwise.
template <bool OUT_BF16, bool ACCUM>
__global__ __launch_bounds__(256)
void gemm_bf16_kernel(const bf16* __restrict__ A, const bf16* __restrict__ Bt,
                      void* __restrict__ Cout, int M, int N, int K) {
    constexpr int LDT = 48;   // 32 cols + 16-element (32-byte) pad, keeps 16B alignment
    __shared__ __attribute__((aligned(16))) bf16 As[2][128 * LDT];
    __shared__ __attribute__((aligned(16))) bf16 Bs[2][128 * LDT];

    const int m0 = blockIdx.y * 128;
    const int n0 = blockIdx.x * 128;
    const int w  = threadIdx.x >> 5;
    const int wm = (w & 3) * 32;     // 32-row strip
    const int wn = (w >> 2) * 64;    // 64-col strip
    const int lane = threadIdx.x & 31;
    const int sel  = lane >> 4;
    const int lcol = lane & 15;

    v8f acc[2][4] = {};

    // stage one 128x32 K-slab of A and Bt into LDS buffer `buf`
    auto issue_tile = [&](int kt, int buf) {
        #pragma unroll
        for (int i = 0; i < 2; ++i) {
            const int c   = threadIdx.x + 256 * i;
            const int row = c >> 2;
            const int cc  = (c & 3) * 8;
            const bf16* ga = &A[(size_t)(m0 + row) * K + kt + cc];
            const bf16* gb = &Bt[(size_t)(n0 + row) * K + kt + cc];
            bf16* la = &As[buf][row * LDT + cc];
            bf16* lb = &Bs[buf][row * LDT + cc];
#if HAVE_ASYNC_LDS
            __builtin_amdgcn_global_load_async_to_lds_b128(
                (as1_v4i*)ga, (as3_v4i*)la, 0, 0);
            __builtin_amdgcn_global_load_async_to_lds_b128(
                (as1_v4i*)gb, (as3_v4i*)lb, 0, 0);
#else
            *reinterpret_cast<v8bf*>(la) = *reinterpret_cast<const v8bf*>(ga);
            *reinterpret_cast<v8bf*>(lb) = *reinterpret_cast<const v8bf*>(gb);
#endif
        }
    };

    const int T = K >> 5;   // K / 32 slabs
    issue_tile(0, 0);
    for (int t = 0; t < T; ++t) {
        const int buf = t & 1;
        wait_async_zero();        // my async loads for buf landed in LDS
        __syncthreads();          // everyone's loads landed; prev compute done
        if (t + 1 < T) issue_tile((t + 1) << 5, buf ^ 1);   // overlap next slab

        v16bf af[2], bfr[4];
        #pragma unroll
        for (int i = 0; i < 2; ++i)
            af[i] = load_frag16x32(&As[buf][(wm + 16 * i) * LDT], LDT);
        #pragma unroll
        for (int j = 0; j < 4; ++j)
            bfr[j] = load_frag16x32(&Bs[buf][(wn + 16 * j) * LDT], LDT);
        #pragma unroll
        for (int i = 0; i < 2; ++i)
            #pragma unroll
            for (int j = 0; j < 4; ++j)
                acc[i][j] = wmma_bf16(af[i], bfr[j], acc[i][j]);
        __syncthreads();          // done reading buf before it is overwritten
    }

    // epilogue: C layout => VGPR r, lanes 0-15: M=r, lanes 16-31: M=r+8
    #pragma unroll
    for (int i = 0; i < 2; ++i)
        #pragma unroll
        for (int j = 0; j < 4; ++j)
            #pragma unroll
            for (int r = 0; r < 8; ++r) {
                const int gm = m0 + wm + 16 * i + r + 8 * sel;
                const int gn = n0 + wn + 16 * j + lcol;
                const size_t idx = (size_t)gm * N + gn;
                if (OUT_BF16) {
                    ((bf16*)Cout)[idx] = (bf16)acc[i][j][r];
                } else {
                    float v = acc[i][j][r];
                    if (ACCUM) v += ((float*)Cout)[idx];
                    ((float*)Cout)[idx] = v;
                }
            }
}

// ---------------- transpose V slice: proj[b,s,COL_V+d] -> vT[b,d,s] ----------------
__global__ __launch_bounds__(256)
void transpose_v_kernel(const bf16* __restrict__ proj, bf16* __restrict__ vT) {
    const int idx = blockIdx.x * 256 + threadIdx.x;   // over 2*128*2048
    const int s = idx & (SEQ - 1);
    const int d = (idx >> 11) & 127;
    const int b = idx >> 18;
    vT[idx] = proj[(size_t)(b * SEQ + s) * NPROJ + COL_V + d];
}

// ---------------- FF activation: ffA = ff * swish(gate) ----------------
__global__ __launch_bounds__(256)
void ff_act_kernel(const bf16* __restrict__ proj, bf16* __restrict__ ffA) {
    const size_t idx = (size_t)blockIdx.x * 256 + threadIdx.x;  // over 4096*8192
    const size_t row = idx >> 13;
    const int    c   = (int)(idx & (FF_INNER - 1));
    const float f = (float)proj[row * NPROJ + COL_FF + c];
    const float g = (float)proj[row * NPROJ + COL_GATE + c];
    ffA[idx] = (bf16)(f * (g / (1.f + __expf(-g))));
}

// ---------------- flash attention (MQA), one wave per 16-query tile ----------------
__global__ __launch_bounds__(32)
void flash_attn_kernel(const bf16* __restrict__ proj, const bf16* __restrict__ vT,
                       const float* __restrict__ bias, bf16* __restrict__ attnA) {
    const int lane = threadIdx.x & 31;
    const int sel  = lane >> 4;
    const int lcol = lane & 15;
    const int q0 = blockIdx.x * 16;
    const int h  = blockIdx.y;
    const int b  = blockIdx.z;

    const bf16*  qbase = proj + ((size_t)(b * SEQ + q0)) * NPROJ + h * DIM_HEAD;
    const bf16*  kbase = proj + (size_t)b * SEQ * NPROJ + COL_K;
    const bf16*  vbase = vT + (size_t)b * DIM_HEAD * SEQ;
    const float* bbase = bias + ((size_t)h * SEQ + q0) * SEQ;

    __shared__ __attribute__((aligned(16))) bf16 Plds[16 * 32];

    v16bf qf[4];
    #pragma unroll
    for (int kt = 0; kt < 4; ++kt) qf[kt] = load_frag16x32(qbase + 32 * kt, NPROJ);

    v8f o[8] = {};
    float rmax[8], rsum[8];
    #pragma unroll
    for (int r = 0; r < 8; ++r) { rmax[r] = -3.0e38f; rsum[r] = 0.f; }

    for (int j0 = 0; j0 < SEQ; j0 += 32) {
        v8f s0 = {}, s1 = {};
        #pragma unroll
        for (int kt = 0; kt < 4; ++kt) {
            v16bf k0 = load_frag16x32(kbase + (size_t)j0 * NPROJ + 32 * kt, NPROJ);
            v16bf k1 = load_frag16x32(kbase + (size_t)(j0 + 16) * NPROJ + 32 * kt, NPROJ);
            s0 = wmma_bf16(qf[kt], k0, s0);
            s1 = wmma_bf16(qf[kt], k1, s1);
        }
        float corr[8];
        #pragma unroll
        for (int r = 0; r < 8; ++r) {
            const int mrow = r + 8 * sel;
            const float b0 = bbase[(size_t)mrow * SEQ + j0 + lcol];
            const float b1 = bbase[(size_t)mrow * SEQ + j0 + 16 + lcol];
            float v0 = s0[r] * Q_SCALE_F + b0;
            float v1 = s1[r] * Q_SCALE_F + b1;
            float m = fmaxf(v0, v1);
            #pragma unroll
            for (int off = 8; off > 0; off >>= 1) m = fmaxf(m, __shfl_xor(m, off));
            const float nm = fmaxf(rmax[r], m);
            const float c  = __expf(rmax[r] - nm);
            const float p0 = __expf(v0 - nm);
            const float p1 = __expf(v1 - nm);
            float ps = p0 + p1;
            #pragma unroll
            for (int off = 8; off > 0; off >>= 1) ps += __shfl_xor(ps, off);
            rsum[r] = rsum[r] * c + ps;
            rmax[r] = nm;
            corr[r] = c;
            Plds[mrow * 32 + lcol]      = (bf16)p0;   // per-wave DS ops are in-order
            Plds[mrow * 32 + lcol + 16] = (bf16)p1;
        }
        #pragma unroll
        for (int n = 0; n < 8; ++n)
            #pragma unroll
            for (int r = 0; r < 8; ++r) o[n][r] *= corr[r];

        v16bf pf = load_frag16x32(Plds, 32);
        #pragma unroll
        for (int n = 0; n < 8; ++n) {
            v16bf vf = load_frag16x32(vbase + (size_t)(n * 16) * SEQ + j0, SEQ);
            o[n] = wmma_bf16(pf, vf, o[n]);
        }
    }

    #pragma unroll
    for (int n = 0; n < 8; ++n)
        #pragma unroll
        for (int r = 0; r < 8; ++r) {
            const int gm = b * SEQ + q0 + r + 8 * sel;
            const int gn = h * DIM_HEAD + n * 16 + lcol;
            attnA[(size_t)gm * ATTN_INNER + gn] = (bf16)(o[n][r] / rsum[r]);
        }
}

// ---------------- host launcher ----------------
static inline size_t align256(size_t x) { return (x + 255) & ~size_t(255); }

extern "C" void kernel_launch(void* const* d_in, const int* in_sizes, int n_in,
                              void* d_out, int out_size, void* d_ws, size_t ws_size,
                              hipStream_t stream) {
    const float* x       = (const float*)d_in[0];
    const float* bias    = (const float*)d_in[1];
    const float* gamma   = (const float*)d_in[2];
    const float* wi      = (const float*)d_in[3];
    const float* attn_wo = (const float*)d_in[4];
    const float* ff_wo   = (const float*)d_in[5];
    float* out = (float*)d_out;

    // workspace carve-up (bf16 buffers)
    char* ws = (char*)d_ws;
    size_t off = 0;
    bf16* wiT     = (bf16*)(ws + off); off = align256(off + (size_t)NPROJ * DIMC * 2);
    bf16* awoT    = (bf16*)(ws + off); off = align256(off + (size_t)DIMC * ATTN_INNER * 2);
    bf16* fwoT    = (bf16*)(ws + off); off = align256(off + (size_t)DIMC * FF_INNER * 2);
    bf16* xnbf    = (bf16*)(ws + off); off = align256(off + (size_t)MROWS * DIMC * 2);
    bf16* projbf  = (bf16*)(ws + off); off = align256(off + (size_t)MROWS * NPROJ * 2);
    bf16* vT      = (bf16*)(ws + off); off = align256(off + (size_t)BATCH * DIM_HEAD * SEQ * 2);
    bf16* attnA   = (bf16*)(ws + off); off = align256(off + (size_t)MROWS * ATTN_INNER * 2);
    bf16* ffA     = (bf16*)(ws + off); off = align256(off + (size_t)MROWS * FF_INNER * 2);
    (void)ws_size; (void)in_sizes; (void)n_in; (void)out_size;

    const dim3 tb(32, 8);
    // weights -> bf16, N-major (B^T)
    transpose_cvt_kernel<<<dim3(NPROJ / 32, DIMC / 32), tb, 0, stream>>>(wi, wiT, DIMC, NPROJ);
    transpose_cvt_kernel<<<dim3(DIMC / 32, ATTN_INNER / 32), tb, 0, stream>>>(attn_wo, awoT, ATTN_INNER, DIMC);
    transpose_cvt_kernel<<<dim3(DIMC / 32, FF_INNER / 32), tb, 0, stream>>>(ff_wo, fwoT, FF_INNER, DIMC);

    // RMSNorm
    rmsnorm_kernel<<<MROWS, 256, 0, stream>>>(x, gamma, xnbf);

    // fused projection: [4096,2048] x [2048,18688] -> bf16
    gemm_bf16_kernel<true, false><<<dim3(NPROJ / 128, MROWS / 128), 256, 0, stream>>>(
        xnbf, wiT, projbf, MROWS, NPROJ, DIMC);

    // V transpose + FF activation
    transpose_v_kernel<<<(BATCH * DIM_HEAD * SEQ) / 256, 256, 0, stream>>>(projbf, vT);
    ff_act_kernel<<<((size_t)MROWS * FF_INNER) / 256, 256, 0, stream>>>(projbf, ffA);

    // flash attention
    flash_attn_kernel<<<dim3(SEQ / 16, HEADS, BATCH), 32, 0, stream>>>(projbf, vT, bias, attnA);

    // attn_out -> d_out, then ff_out accumulated into d_out
    gemm_bf16_kernel<false, false><<<dim3(DIMC / 128, MROWS / 128), 256, 0, stream>>>(
        attnA, awoT, out, MROWS, DIMC, ATTN_INNER);
    gemm_bf16_kernel<false, true><<<dim3(DIMC / 128, MROWS / 128), 256, 0, stream>>>(
        ffA, fwoT, out, MROWS, DIMC, FF_INNER);
}
